// OfficialMambaBlock_9354438771059
// MI455X (gfx1250) — compile-verified, matrix operand order checked by reading, not run on hardware
//
#include <hip/hip_runtime.h>

// ---------------------------------------------------------------------------
// Mamba block for MI455X (gfx1250, wave32, WMMA bf16 16x16x32 w/ f32 accum)
// ---------------------------------------------------------------------------

#define D_MODEL 2048
#define D_STATE 16
#define KER     4
#define DT_RANK 128          // D_MODEL / 16
#define B_SZ    2
#define SEQ     1024
#define NTOK    (B_SZ * SEQ) // 2048 tokens
#define XPROJ_N 160          // DT_RANK + 2*D_STATE
#define XPROJ_P 192          // padded to a multiple of 64 for the GEMM tile

typedef __attribute__((ext_vector_type(16))) __bf16 v16bf;
typedef __attribute__((ext_vector_type(8)))  __bf16 v8bf;
typedef __attribute__((ext_vector_type(8)))  float  v8f;

// ---------------------------------------------------------------------------
// WMMA helper
// ---------------------------------------------------------------------------
__device__ __forceinline__ v8f wmma_bf16(v16bf a, v16bf b, v8f c) {
  // (neg_a, A, neg_b, B, c_mod, C, reuse_a, reuse_b)
  return __builtin_amdgcn_wmma_f32_16x16x32_bf16(false, a, false, b,
                                                 (short)0, c, false, false);
}

// Load one operand fragment for the 16x32 bf16 A/B layout.
// lane r (0..15) holds matrix row/col r; lane half selects the K-subblock.
// Element e (0..7)  -> K = koff + e        (koff already includes half*8)
// Element e (8..15) -> K = koff + 16 + (e-8)
__device__ __forceinline__ v16bf load_frag(const __bf16* __restrict__ p) {
  v8bf lo = *(const v8bf*)(p);
  v8bf hi = *(const v8bf*)(p + 16);
  v16bf out;
#pragma unroll
  for (int i = 0; i < 8; ++i) { out[i] = lo[i]; out[i + 8] = hi[i]; }
  return out;
}

// ---------------------------------------------------------------------------
// GEMM: C[M,N] = A[M,K] * W[N,K]^T   (A,W bf16 row-major over K; C fp32)
// One wave (32 threads) computes a 64x64 output macro-tile:
//   4 A-tiles (M) x 4 B-tiles (N) = 16 WMMA per K-step of 32.
// K-loop is NOT unrolled and tile loops have compile-time bounds so the
// 16 f32 accumulator tiles (128 VGPRs) stay resident in registers.
// Requires M % 64 == 0, N % 64 == 0, K % 32 == 0.
// ---------------------------------------------------------------------------
__global__ void __launch_bounds__(32, 1)
gemm_bf16_wmma(const __bf16* __restrict__ A,
               const __bf16* __restrict__ W,
               float* __restrict__ C,
               int M, int N, int K) {
  const int lane = threadIdx.x;       // blockDim.x == 32
  const int r    = lane & 15;
  const int half = lane >> 4;
  const int n0   = blockIdx.x * 64;
  const int m0   = blockIdx.y * 64;

  v8f acc[4][4];
#pragma unroll
  for (int mi = 0; mi < 4; ++mi)
#pragma unroll
    for (int j = 0; j < 4; ++j) {
      v8f z = {0.f, 0.f, 0.f, 0.f, 0.f, 0.f, 0.f, 0.f};
      acc[mi][j] = z;
    }

  // Row base pointers (advance by 32 bf16 per K-step).
  const __bf16* ap0 = A + (size_t)(m0 + r) * K + half * 8;
  const __bf16* ap1 = ap0 + (size_t)16 * K;
  const __bf16* ap2 = ap0 + (size_t)32 * K;
  const __bf16* ap3 = ap0 + (size_t)48 * K;
  const __bf16* wp0 = W + (size_t)(n0 + r) * K + half * 8;
  const __bf16* wp1 = wp0 + (size_t)16 * K;
  const __bf16* wp2 = wp0 + (size_t)32 * K;
  const __bf16* wp3 = wp0 + (size_t)48 * K;

#pragma unroll 1
  for (int k0 = 0; k0 < K; k0 += 32) {
    v16bf a0 = load_frag(ap0); ap0 += 32;
    v16bf a1 = load_frag(ap1); ap1 += 32;
    v16bf a2 = load_frag(ap2); ap2 += 32;
    v16bf a3 = load_frag(ap3); ap3 += 32;

    v16bf b0 = load_frag(wp0); wp0 += 32;
    acc[0][0] = wmma_bf16(a0, b0, acc[0][0]);
    acc[1][0] = wmma_bf16(a1, b0, acc[1][0]);
    acc[2][0] = wmma_bf16(a2, b0, acc[2][0]);
    acc[3][0] = wmma_bf16(a3, b0, acc[3][0]);

    v16bf b1 = load_frag(wp1); wp1 += 32;
    acc[0][1] = wmma_bf16(a0, b1, acc[0][1]);
    acc[1][1] = wmma_bf16(a1, b1, acc[1][1]);
    acc[2][1] = wmma_bf16(a2, b1, acc[2][1]);
    acc[3][1] = wmma_bf16(a3, b1, acc[3][1]);

    v16bf b2 = load_frag(wp2); wp2 += 32;
    acc[0][2] = wmma_bf16(a0, b2, acc[0][2]);
    acc[1][2] = wmma_bf16(a1, b2, acc[1][2]);
    acc[2][2] = wmma_bf16(a2, b2, acc[2][2]);
    acc[3][2] = wmma_bf16(a3, b2, acc[3][2]);

    v16bf b3 = load_frag(wp3); wp3 += 32;
    acc[0][3] = wmma_bf16(a0, b3, acc[0][3]);
    acc[1][3] = wmma_bf16(a1, b3, acc[1][3]);
    acc[2][3] = wmma_bf16(a2, b3, acc[2][3]);
    acc[3][3] = wmma_bf16(a3, b3, acc[3][3]);
  }

  // D layout: VGPR rr holds M = rr + 8*half, N = (lane&15); store fp32.
#pragma unroll
  for (int mi = 0; mi < 4; ++mi)
#pragma unroll
    for (int j = 0; j < 4; ++j)
#pragma unroll
      for (int rr = 0; rr < 8; ++rr)
        C[(size_t)(m0 + mi * 16 + half * 8 + rr) * N + n0 + j * 16 + r] =
            acc[mi][j][rr];
}

// ---------------------------------------------------------------------------
// fp32 -> bf16 conversion
// ---------------------------------------------------------------------------
__global__ void cvt_f32_bf16(const float* __restrict__ src,
                             __bf16* __restrict__ dst, int n) {
  int i = blockIdx.x * blockDim.x + threadIdx.x;
  if (i < n) dst[i] = (__bf16)src[i];
}

// fp32 -> bf16 with row zero-padding: dst has nrows_pad rows of K cols,
// src has nrows rows; rows >= nrows are zero-filled.
__global__ void cvt_f32_bf16_padrows(const float* __restrict__ src,
                                     __bf16* __restrict__ dst,
                                     int nrows, int nrows_pad, int K) {
  int i = blockIdx.x * blockDim.x + threadIdx.x;
  if (i >= nrows_pad * K) return;
  int row = i / K;
  dst[i] = (row < nrows) ? (__bf16)src[i] : (__bf16)0.f;
}

// ---------------------------------------------------------------------------
// Depthwise causal conv (K=4) + SiLU.  xz = [u_pre | z] per token (4096 cols)
// ---------------------------------------------------------------------------
__global__ void conv_silu(const float* __restrict__ xz,
                          const float* __restrict__ cw,
                          const float* __restrict__ cb,
                          float* __restrict__ u,
                          __bf16* __restrict__ ubf) {
  int idx = blockIdx.x * blockDim.x + threadIdx.x;  // NTOK * D_MODEL
  if (idx >= NTOK * D_MODEL) return;
  int d   = idx & (D_MODEL - 1);
  int tok = idx >> 11;
  int t   = tok & (SEQ - 1);
  float acc = cb[d];
#pragma unroll
  for (int k = 0; k < KER; ++k) {
    int tt = t - (KER - 1) + k;
    if (tt >= 0)
      acc += cw[d * KER + k] * xz[(size_t)(tok - (KER - 1) + k) * (2 * D_MODEL) + d];
  }
  float s = acc / (1.f + __expf(-acc));   // silu
  u[idx]   = s;
  ubf[idx] = (__bf16)s;
}

// ---------------------------------------------------------------------------
// Slice dt_r (first 128 cols of x_dbl, padded stride) to bf16
// ---------------------------------------------------------------------------
__global__ void slice_dtr(const float* __restrict__ xdbl,
                          __bf16* __restrict__ dtrbf) {
  int i = blockIdx.x * blockDim.x + threadIdx.x;  // NTOK * DT_RANK
  if (i >= NTOK * DT_RANK) return;
  int t = i >> 7, c = i & (DT_RANK - 1);
  dtrbf[i] = (__bf16)xdbl[(size_t)t * XPROJ_P + c];
}

// ---------------------------------------------------------------------------
// dt = softplus(dt_pre + b_dt), in place
// ---------------------------------------------------------------------------
__global__ void softplus_dt(float* __restrict__ dtpre,
                            const float* __restrict__ b_dt) {
  int i = blockIdx.x * blockDim.x + threadIdx.x;  // NTOK * D_MODEL
  if (i >= NTOK * D_MODEL) return;
  int d = i & (D_MODEL - 1);
  float x = dtpre[i] + b_dt[d];
  dtpre[i] = (x > 20.f) ? x : log1pf(__expf(x));
}

// ---------------------------------------------------------------------------
// Selective scan: one thread per (b,d) channel, 16 states in registers,
// sequential over SEQ.  B/C rows are lane-uniform (broadcast), dt/u/z
// coalesced across d.  Emits y directly in bf16 for the output GEMM.
// ---------------------------------------------------------------------------
__global__ void scan_kernel(const float* __restrict__ dt,
                            const float* __restrict__ u,
                            const float* __restrict__ xdbl,
                            const float* __restrict__ A_log,
                            const float* __restrict__ Dskip,
                            const float* __restrict__ xz,
                            __bf16* __restrict__ ybf) {
  int gid = blockIdx.x * blockDim.x + threadIdx.x;  // B_SZ * D_MODEL
  if (gid >= B_SZ * D_MODEL) return;
  int d = gid & (D_MODEL - 1);
  int b = gid >> 11;

  float a[D_STATE], h[D_STATE];
#pragma unroll
  for (int n = 0; n < D_STATE; ++n) {
    a[n] = -__expf(A_log[d * D_STATE + n]);
    h[n] = 0.f;
  }
  const float dsk = Dskip[d];

  for (int t = 0; t < SEQ; ++t) {
    int tok = b * SEQ + t;
    float dtv = dt[(size_t)tok * D_MODEL + d];
    float uv  = u[(size_t)tok * D_MODEL + d];
    const float* Brow = xdbl + (size_t)tok * XPROJ_P + DT_RANK;
    const float* Crow = Brow + D_STATE;
    float y = 0.f;
#pragma unroll
    for (int n = 0; n < D_STATE; ++n) {
      float dA = __expf(dtv * a[n]);
      h[n] = dA * h[n] + dtv * Brow[n] * uv;
      y += h[n] * Crow[n];
    }
    float zv  = xz[(size_t)tok * (2 * D_MODEL) + D_MODEL + d];
    float out = (y + uv * dsk) * (zv / (1.f + __expf(-zv)));
    ybf[(size_t)tok * D_MODEL + d] = (__bf16)out;
  }
}

// ---------------------------------------------------------------------------
// Host-side launch
// ---------------------------------------------------------------------------
static inline size_t align256(size_t x) { return (x + 255) & ~(size_t)255; }

extern "C" void kernel_launch(void* const* d_in, const int* in_sizes, int n_in,
                              void* d_out, int out_size, void* d_ws, size_t ws_size,
                              hipStream_t stream) {
  (void)in_sizes; (void)n_in; (void)out_size; (void)ws_size;

  const float* x      = (const float*)d_in[0]; // (2,1024,2048)
  const float* W_in   = (const float*)d_in[1]; // (4096,2048)
  const float* conv_w = (const float*)d_in[2]; // (2048,4)
  const float* conv_b = (const float*)d_in[3]; // (2048,)
  const float* W_xp   = (const float*)d_in[4]; // (160,2048)
  const float* W_dt   = (const float*)d_in[5]; // (2048,128)
  const float* b_dt   = (const float*)d_in[6]; // (2048,)
  const float* A_log  = (const float*)d_in[7]; // (2048,16)
  const float* Dskip  = (const float*)d_in[8]; // (2048,)
  const float* W_out  = (const float*)d_in[9]; // (2048,2048)
  float* out = (float*)d_out;                  // (2,1024,2048)

  // ---- workspace carve-up (bytes) ----
  char* ws = (char*)d_ws;
  size_t off = 0;
  auto take = [&](size_t bytes) { char* p = ws + off; off += align256(bytes); return p; };

  __bf16* xbf    = (__bf16*)take((size_t)NTOK * D_MODEL * 2);          // 8 MB
  __bf16* Winbf  = (__bf16*)take((size_t)2 * D_MODEL * D_MODEL * 2);   // 16 MB
  float*  xz     = (float*) take((size_t)NTOK * 2 * D_MODEL * 4);      // 32 MB
  float*  u      = (float*) take((size_t)NTOK * D_MODEL * 4);          // 16 MB
  __bf16* ubf    = (__bf16*)take((size_t)NTOK * D_MODEL * 2);          // 8 MB
  __bf16* Wxpbf  = (__bf16*)take((size_t)XPROJ_P * D_MODEL * 2);       // padded
  float*  xdbl   = (float*) take((size_t)NTOK * XPROJ_P * 4);          // padded
  __bf16* dtrbf  = (__bf16*)take((size_t)NTOK * DT_RANK * 2);
  __bf16* Wdtbf  = (__bf16*)take((size_t)D_MODEL * DT_RANK * 2);
  float*  dtbuf  = (float*) take((size_t)NTOK * D_MODEL * 4);          // 16 MB
  __bf16* ybf    = (__bf16*)take((size_t)NTOK * D_MODEL * 2);          // 8 MB
  __bf16* Woutbf = (__bf16*)take((size_t)D_MODEL * D_MODEL * 2);       // 8 MB

  const int T = 256;
  auto blocks = [&](int n) { return (n + T - 1) / T; };

  // 1) convert x and W_in, GEMM xz = x @ W_in^T  (M=2048,N=4096,K=2048)
  cvt_f32_bf16<<<blocks(NTOK * D_MODEL), T, 0, stream>>>(x, xbf, NTOK * D_MODEL);
  cvt_f32_bf16<<<blocks(2 * D_MODEL * D_MODEL), T, 0, stream>>>(W_in, Winbf, 2 * D_MODEL * D_MODEL);
  gemm_bf16_wmma<<<dim3((2 * D_MODEL) / 64, NTOK / 64), 32, 0, stream>>>(
      xbf, Winbf, xz, NTOK, 2 * D_MODEL, D_MODEL);

  // 2) depthwise causal conv + silu -> u (f32 + bf16)
  conv_silu<<<blocks(NTOK * D_MODEL), T, 0, stream>>>(xz, conv_w, conv_b, u, ubf);

  // 3) x_dbl = u @ W_xproj^T  (M=2048, N=192 padded, K=2048)
  cvt_f32_bf16_padrows<<<blocks(XPROJ_P * D_MODEL), T, 0, stream>>>(
      W_xp, Wxpbf, XPROJ_N, XPROJ_P, D_MODEL);
  gemm_bf16_wmma<<<dim3(XPROJ_P / 64, NTOK / 64), 32, 0, stream>>>(
      ubf, Wxpbf, xdbl, NTOK, XPROJ_P, D_MODEL);

  // 4) dt_pre = dt_r @ W_dt^T  (M=2048,N=2048,K=128), then softplus(+b_dt)
  slice_dtr<<<blocks(NTOK * DT_RANK), T, 0, stream>>>(xdbl, dtrbf);
  cvt_f32_bf16<<<blocks(D_MODEL * DT_RANK), T, 0, stream>>>(W_dt, Wdtbf, D_MODEL * DT_RANK);
  gemm_bf16_wmma<<<dim3(D_MODEL / 64, NTOK / 64), 32, 0, stream>>>(
      dtrbf, Wdtbf, dtbuf, NTOK, D_MODEL, DT_RANK);
  softplus_dt<<<blocks(NTOK * D_MODEL), T, 0, stream>>>(dtbuf, b_dt);

  // 5) selective scan -> y (bf16), fused with D-skip and silu(z) gate
  scan_kernel<<<blocks(B_SZ * D_MODEL), T, 0, stream>>>(
      dtbuf, u, xdbl, A_log, Dskip, xz, ybf);

  // 6) out = y @ W_out^T  (M=2048,N=2048,K=2048) -> fp32 d_out
  cvt_f32_bf16<<<blocks(D_MODEL * D_MODEL), T, 0, stream>>>(W_out, Woutbf, D_MODEL * D_MODEL);
  gemm_bf16_wmma<<<dim3(D_MODEL / 64, NTOK / 64), 32, 0, stream>>>(
      ybf, Woutbf, out, NTOK, D_MODEL, D_MODEL);
}
